// NeuralODEModel_21715354648710
// MI455X (gfx1250) — matrix operand order repaired and checked by reading, not compiled, not run on hardware
//
#include <hip/hip_runtime.h>

typedef __attribute__((ext_vector_type(16))) _Float16 v16h;
typedef __attribute__((ext_vector_type(8)))  _Float16 v8h;
typedef __attribute__((ext_vector_type(2)))  _Float16 v2h;
typedef __attribute__((ext_vector_type(8)))  float    v8f;

#define DS_SYNC() asm volatile("s_wait_dscnt 0" ::: "memory")

// elementwise over the two 16x16 C-fragments (8 f32 per lane each)
#define EW(stmt) { _Pragma("unroll") for (int ct = 0; ct < 2; ++ct) { \
                   _Pragma("unroll") for (int r = 0; r < 8; ++r) { stmt; } } }

#define LOG2E   1.4426950408889634f
#define LN2     0.6931471805599453f

__device__ __forceinline__ float fast_tanh(float x) {
#if __has_builtin(__builtin_amdgcn_tanhf)
  return __builtin_amdgcn_tanhf(x);                       // V_TANH_F32
#elif __has_builtin(__builtin_amdgcn_tanh_f32)
  return __builtin_amdgcn_tanh_f32(x);
#else
  float e = __builtin_amdgcn_exp2f(x * (2.0f * LOG2E));   // exp(2x)
  return 1.0f - 2.0f * __builtin_amdgcn_rcpf(e + 1.0f);
#endif
}

__device__ __forceinline__ float softplus_f(float x) {    // log1p(exp(x)), stable
  float z = __builtin_amdgcn_exp2f(-fabsf(x) * LOG2E);
  return fmaxf(x, 0.f) + LN2 * __builtin_amdgcn_logf(1.f + z);
}

// A-fragment (16x32 f16, MxK): lane L<16 holds row M=L, K = {kb..kb+7, kb+16..kb+23};
// lane L>=16 holds row M=L-16, K = {kb+8..kb+15, kb+24..kb+31}. Row-major LDS buffer.
__device__ __forceinline__ v16h load_a_frag(const _Float16* buf, int strideh, int lane, int kb) {
  int row  = lane & 15;
  int koff = (lane >> 4) * 8;
  const v8h* p0 = (const v8h*)(buf + row * strideh + kb + koff);
  const v8h* p1 = (const v8h*)(buf + row * strideh + kb + 16 + koff);
  v8h lo = *p0, hi = *p1;
  return __builtin_shufflevector(lo, hi, 0,1,2,3,4,5,6,7,8,9,10,11,12,13,14,15);
}

// B-fragment (32x16 f16, KxN): lane holds column N = ntile*16 + (lane&15),
// K = {kb+8h..kb+8h+7, kb+16+8h..kb+23+8h}, h = (lane>=16). Weights stored transposed
// in LDS as [N][K] so each 8 K-values are contiguous.
__device__ __forceinline__ v16h load_b_frag(const _Float16* buf, int strideh, int lane, int kb, int ntile) {
  int c    = ntile * 16 + (lane & 15);
  int koff = (lane >> 4) * 8;
  const v8h* p = (const v8h*)(buf + c * strideh + kb + koff);
  v8h lo = p[0], hi = p[2];
  return __builtin_shufflevector(lo, hi, 0,1,2,3,4,5,6,7,8,9,10,11,12,13,14,15);
}

__global__ __launch_bounds__(256) void node_fused_kernel(
    const float* __restrict__ tg,   const float* __restrict__ sf,
    const float* __restrict__ injt, const float* __restrict__ injd,
    const float* __restrict__ We,   const float* __restrict__ be,
    const float* __restrict__ W0,   const float* __restrict__ b0,
    const float* __restrict__ decayp,
    const float* __restrict__ W1,   const float* __restrict__ b1,
    const float* __restrict__ W2,   const float* __restrict__ b2,
    const float* __restrict__ Wr,   const float* __restrict__ br,
    const float* __restrict__ Wc,   const float* __restrict__ bc,
    float* __restrict__ out)
{
  // weights transposed to [out][in] f16; W1 K padded 67 -> 96, with bias b1 at K=67
  __shared__ alignas(16) _Float16 sW1t[64][96];   // 12 KB
  __shared__ alignas(16) _Float16 sW2t[32][64];   //  4 KB
  __shared__ alignas(16) _Float16 sW0t[32][32];   //  2 KB
  __shared__ alignas(16) _Float16 sX[8][16*96];   // 24 KB: per-wave ODE input [h(32) se(32) dose sin cos 1 pad]
  __shared__ alignas(16) _Float16 sU[8][16*64];   // 16 KB: per-wave tanh acts (also reused as f32 h-stage)

  const int tid = threadIdx.x;
  const int wv  = tid >> 5;
  const int L   = tid & 31;
  const int lc  = L & 15;
  const int hg  = L >> 4;                  // lane half-group
  const int rowBase = blockIdx.x * 128 + wv * 16;
  const int myrow   = rowBase + lc;

  // ---- one-time weight prep (f32 -> f16, transpose, bias row at K=67) ----
  for (int idx = tid; idx < 64*96; idx += 256) {
    int n = idx / 96, k = idx - n*96;
    float w = (k < 67) ? W1[k*64 + n] : ((k == 67) ? b1[n] : 0.f);
    sW1t[n][k] = (_Float16)w;
  }
  for (int idx = tid; idx < 32*64; idx += 256) {
    int n = idx >> 6, k = idx & 63;
    sW2t[n][k] = (_Float16)W2[k*32 + n];
  }
  for (int idx = tid; idx < 32*32; idx += 256) {
    int n = idx >> 5, k = idx & 31;
    sW0t[n][k] = (_Float16)W0[k*32 + n];
  }

  _Float16* X = &sX[wv][0];
  _Float16* U = &sU[wv][0];

  // zero the pad region K=64..95; constant 1.0 at K=67 (bias column)
  for (int idx = L; idx < 16*32; idx += 32) {
    int r = idx >> 5, c = idx & 31;
    X[r*96 + 64 + c] = (c == 3) ? (_Float16)1.f : (_Float16)0.f;
  }

  // ---- per-row constants ----
  float itv[10], idv[10];
  #pragma unroll
  for (int j = 0; j < 10; ++j) {
    itv[j] = injt[(size_t)myrow*10 + j];
    idv[j] = injd[(size_t)myrow*10 + j];
  }
  const float invp   = 1.0f / (itv[1] - itv[0]);   // 1/period (revolutions for v_sin/v_cos)
  const float adecay = fabsf(decayp[0]);
  const float br0    = br[0];

  float bias2[2], b0v[2];
  #pragma unroll
  for (int n = 0; n < 2; ++n) bias2[n] = b2[n*16 + lc];
  #pragma unroll
  for (int n = 0; n < 2; ++n) b0v[n]  = b0[n*16 + lc];

  // ---- static embedding se = relu(sf@We+be), init_cp, write se into X[K=32..63] ----
  float icp;
  {
    float sfv[7];
    #pragma unroll
    for (int i = 0; i < 7; ++i) sfv[i] = sf[(size_t)myrow*7 + i];
    float p = 0.f;
    #pragma unroll
    for (int j = 0; j < 16; ++j) {
      int c = hg*16 + j;                    // lanes 0..31 cover rows 0..15 x cols 0..31
      float s = be[c];
      #pragma unroll
      for (int i = 0; i < 7; ++i) s += sfv[i] * We[i*32 + c];
      s = fmaxf(s, 0.f);
      X[lc*96 + 32 + c] = (_Float16)s;
      p += s * Wc[c];
    }
    p += __shfl_xor(p, 16);                 // full 32-col dot for row lc
    icp = softplus_f(p + bc[0]);
  }

  __syncthreads();                          // weights + se visible

  // ---- W1/W2 B-fragments register-resident for the whole integration ----
  v16h B1f[3][4], B2f[2][2];
  #pragma unroll
  for (int kc = 0; kc < 3; ++kc)
    #pragma unroll
    for (int n = 0; n < 4; ++n)
      B1f[kc][n] = load_b_frag(&sW1t[0][0], 96, L, kc*32, n);
  #pragma unroll
  for (int kc = 0; kc < 2; ++kc)
    #pragma unroll
    for (int n = 0; n < 2; ++n)
      B2f[kc][n] = load_b_frag(&sW2t[0][0], 64, L, kc*32, n);

  // ---- h0 = se @ W0 + b0 via WMMA (A = se region of X) ----
  v8f y[2];
  {
    v16h a = load_a_frag(X, 96, L, 32);
    #pragma unroll
    for (int ct = 0; ct < 2; ++ct) {
      v16h bw = load_b_frag(&sW0t[0][0], 32, L, 0, ct);
      v8f acc;
      #pragma unroll
      for (int r = 0; r < 8; ++r) acc[r] = b0v[ct];
      acc = __builtin_amdgcn_wmma_f32_16x16x32_f16(false, a, false, bw, (short)0, acc, false, false);
      y[ct] = acc;
    }
  }

  // ---- ODE RHS: k = tanh([h,se,dose,te,1] @ W1ext) @ W2 + b2 - |d|*h ----
  auto rhs = [&](float tc, const v8f yy[2], v8f kk[2]) {
    #pragma unroll
    for (int ct = 0; ct < 2; ++ct)
      #pragma unroll
      for (int r = 0; r < 8; ++r)
        X[(r + hg*8)*96 + ct*16 + lc] = (_Float16)yy[ct][r];   // h into K=0..31

    float dose = 0.f;
    #pragma unroll
    for (int j = 0; j < 10; ++j) {
      float d = tc - itv[j];
      // exp(-d^2/(2*sigma^2)) = exp2(d^2 * (-5e5*log2e)), sigma=1e-3
      dose = fmaf(idv[j], __builtin_amdgcn_exp2f(d*d * -721347.52f), dose);
    }
    float rev = tc * invp;                                  // hardware sin/cos take revolutions
    float sn = __builtin_amdgcn_sinf(rev);
    float cs = __builtin_amdgcn_cosf(rev);
    if (hg == 0) {
      v2h ds2; ds2[0] = (_Float16)dose; ds2[1] = (_Float16)sn;
      *(v2h*)(X + lc*96 + 64) = ds2;                        // one b32 store
      X[lc*96 + 66] = (_Float16)cs;                         // K=67 stays 1.0
    }
    DS_SYNC();

    v8f acc[4] = {};                                        // SRC2 inline 0 on first chunk
    #pragma unroll
    for (int kc = 0; kc < 3; ++kc) {
      v16h a = load_a_frag(X, 96, L, kc*32);
      #pragma unroll
      for (int n = 0; n < 4; ++n)
        acc[n] = __builtin_amdgcn_wmma_f32_16x16x32_f16(false, a, false, B1f[kc][n], (short)0, acc[n], false, false);
    }

    #pragma unroll
    for (int n = 0; n < 4; ++n)
      #pragma unroll
      for (int r = 0; r < 8; ++r)
        U[(r + hg*8)*64 + n*16 + lc] = (_Float16)fast_tanh(acc[n][r]);
    DS_SYNC();

    v8f a2[2] = {};
    #pragma unroll
    for (int kc = 0; kc < 2; ++kc) {
      v16h a = load_a_frag(U, 64, L, kc*32);
      #pragma unroll
      for (int n = 0; n < 2; ++n)
        a2[n] = __builtin_amdgcn_wmma_f32_16x16x32_f16(false, a, false, B2f[kc][n], (short)0, a2[n], false, false);
    }
    #pragma unroll
    for (int ct = 0; ct < 2; ++ct)
      #pragma unroll
      for (int r = 0; r < 8; ++r)
        kk[ct][r] = (a2[ct][r] + bias2[ct]) - adecay * yy[ct][r];
    DS_SYNC();
  };

  // ---- readout: out[row,ti] = softplus(h@Wr+br) + init_cp ----
  auto write_out = [&](int ti) {
    float* H = reinterpret_cast<float*>(U);   // reuse U slot as f32 [16][32]
    DS_SYNC();
    #pragma unroll
    for (int ct = 0; ct < 2; ++ct)
      #pragma unroll
      for (int r = 0; r < 8; ++r)
        H[(r + hg*8)*32 + ct*16 + lc] = y[ct][r];
    DS_SYNC();
    if (hg == 0) {
      float a = 0.f;
      const float* hr = H + lc*32;
      #pragma unroll
      for (int c = 0; c < 32; ++c) a += hr[c] * Wr[c];
      out[(size_t)myrow*17 + ti] = softplus_f(a + br0) + icp;
    }
    DS_SYNC();
  };

  // ---- fixed-step Dormand-Prince over 16 intervals x 2 substeps ----
  write_out(0);
  for (int iv = 0; iv < 16; ++iv) {
    float t0 = tg[iv], t1 = tg[iv+1];
    float dt = (t1 - t0) * 0.5f;
    for (int sub = 0; sub < 2; ++sub) {
      float ts = t0 + (float)sub * dt;
      v8f k1[2], k2[2], k3[2], k4[2], k5[2], k6[2], yt[2];
      rhs(ts, y, k1);
      EW(yt[ct][r] = y[ct][r] + dt*0.2f*k1[ct][r]);
      rhs(ts + 0.2f*dt, yt, k2);
      EW(yt[ct][r] = y[ct][r] + dt*((3.f/40.f)*k1[ct][r] + (9.f/40.f)*k2[ct][r]));
      rhs(ts + 0.3f*dt, yt, k3);
      EW(yt[ct][r] = y[ct][r] + dt*((44.f/45.f)*k1[ct][r] - (56.f/15.f)*k2[ct][r]
                                  + (32.f/9.f)*k3[ct][r]));
      rhs(ts + 0.8f*dt, yt, k4);
      EW(yt[ct][r] = y[ct][r] + dt*((19372.f/6561.f)*k1[ct][r] - (25360.f/2187.f)*k2[ct][r]
                                  + (64448.f/6561.f)*k3[ct][r] - (212.f/729.f)*k4[ct][r]));
      rhs(ts + (8.f/9.f)*dt, yt, k5);
      EW(yt[ct][r] = y[ct][r] + dt*((9017.f/3168.f)*k1[ct][r] - (355.f/33.f)*k2[ct][r]
                                  + (46732.f/5247.f)*k3[ct][r] + (49.f/176.f)*k4[ct][r]
                                  - (5103.f/18656.f)*k5[ct][r]));
      rhs(ts + dt, yt, k6);
      EW(y[ct][r] = y[ct][r] + dt*((35.f/384.f)*k1[ct][r] + (500.f/1113.f)*k3[ct][r]
                                 + (125.f/192.f)*k4[ct][r] - (2187.f/6784.f)*k5[ct][r]
                                 + (11.f/84.f)*k6[ct][r]));
    }
    write_out(iv + 1);
  }
}

extern "C" void kernel_launch(void* const* d_in, const int* in_sizes, int n_in,
                              void* d_out, int out_size, void* d_ws, size_t ws_size,
                              hipStream_t stream) {
  const float* tg   = (const float*)d_in[0];
  const float* sf   = (const float*)d_in[1];
  const float* injt = (const float*)d_in[2];
  const float* injd = (const float*)d_in[3];
  const float* We   = (const float*)d_in[4];
  const float* be   = (const float*)d_in[5];
  const float* W0   = (const float*)d_in[6];
  const float* b0   = (const float*)d_in[7];
  const float* dp   = (const float*)d_in[8];
  const float* W1   = (const float*)d_in[9];
  const float* b1   = (const float*)d_in[10];
  const float* W2   = (const float*)d_in[11];
  const float* b2   = (const float*)d_in[12];
  const float* Wr   = (const float*)d_in[13];
  const float* br   = (const float*)d_in[14];
  const float* Wc   = (const float*)d_in[15];
  const float* bc   = (const float*)d_in[16];

  const int B = in_sizes[1] / 7;          // static_features is [B,7]
  const int blocks = B / 128;             // 8 waves/block x 16 rows/wave

  node_fused_kernel<<<blocks, 256, 0, stream>>>(
      tg, sf, injt, injd, We, be, W0, b0, dp, W1, b1, W2, b2, Wr, br, Wc, bc,
      (float*)d_out);
}
